// LinkPredictor_88364657148085
// MI455X (gfx1250) — matrix operand order repaired and checked by reading, not compiled, not run on hardware
//
#include <hip/hip_runtime.h>

typedef __attribute__((ext_vector_type(16))) __bf16 v16bf;
typedef __attribute__((ext_vector_type(8)))  float v8f;

#define NA_  50000
#define NB_  50000
#define F0_  128
#define H_   256
#define O_   16
#define E_   1000000
#define L_   100000
#define ORD_NEG_INF 0x007FFFFFu

__device__ __forceinline__ unsigned short f2bf(float f) {
  unsigned u = __float_as_uint(f);
  u += 0x7FFFu + ((u >> 16) & 1u);          // round-to-nearest-even
  return (unsigned short)(u >> 16);
}
__device__ __forceinline__ unsigned f2ord(float f) {
  unsigned u = __float_as_uint(f);
  return (u >> 31) ? ~u : (u | 0x80000000u);
}
__device__ __forceinline__ float ord2f(unsigned o) {
  return (o >> 31) ? __uint_as_float(o & 0x7FFFFFFFu) : __uint_as_float(~o);
}

union AFrag { v16bf v; uint4 q[2]; unsigned short u[16]; };
union CFrag { v8f v; float f[8]; };

// Elementwise fp32 -> bf16 cast, 2 elems/thread (n must be even).
__global__ __launch_bounds__(256) void cast_bf16_k(
    const float* __restrict__ in, unsigned* __restrict__ out, size_t n2) {
  size_t i = (size_t)blockIdx.x * blockDim.x + threadIdx.x;
  size_t stride = (size_t)gridDim.x * blockDim.x;
  for (; i < n2; i += stride) {
    float2 f = ((const float2*)in)[i];
    out[i] = (unsigned)f2bf(f.x) | ((unsigned)f2bf(f.y) << 16);
  }
}

// W[K,N] fp32 -> Wt[N,K] bf16 (tiny matrices, at most 256x256).
__global__ __launch_bounds__(256) void transpose_bf16_k(
    const float* __restrict__ W, unsigned short* __restrict__ Wt, int K, int N) {
  int idx = blockIdx.x * blockDim.x + threadIdx.x;
  if (idx >= K * N) return;
  int k = idx / N, n = idx - k * N;
  Wt[(size_t)n * K + k] = f2bf(W[idx]);
}

// C[M,N] = A[M,K] @ W[K,N]; A bf16 row-major, Wt bf16 = W transposed [N,K].
// One wave computes a 16-row strip x TN 16-col tiles; A fragment reused TN times.
// Per 32-K step: 2 b128 loads (A) + TN*2 b128 loads (B, all issued before the
// WMMAs so waits can be partial) + TN WMMAs.
template <int TN>
__global__ __launch_bounds__(256) void gat_wmma_gemm(
    const unsigned short* __restrict__ A, const unsigned short* __restrict__ Wt,
    float* __restrict__ C, int M, int K, int N) {
  int wid  = (int)((blockIdx.x * blockDim.x + threadIdx.x) >> 5);
  int lane = threadIdx.x & 31;
  int groups = (N >> 4) / TN;
  if (wid >= (M >> 4) * groups) return;        // wave-uniform: EXEC stays full
  int tm = wid / groups, tn0 = (wid % groups) * TN;
  int half = lane >> 4, l15 = lane & 15;

  CFrag c[TN];
  #pragma unroll
  for (int t = 0; t < TN; t++)
    #pragma unroll
    for (int i = 0; i < 8; i++) c[t].f[i] = 0.0f;

  const unsigned short* arow = A + (size_t)(tm * 16 + l15) * K;
  const unsigned short* brow[TN];
  #pragma unroll
  for (int t = 0; t < TN; t++)
    brow[t] = Wt + (size_t)((tn0 + t) * 16 + l15) * K;

  for (int k0 = 0; k0 < K; k0 += 32) {
    if (k0 + 32 < K) __builtin_prefetch(arow + k0 + 32, 0, 1);  // global_prefetch_b8
    // A 16x32 bf16 frag: lane<16 row=l15 K={0..7,16..23}; lane>=16 K={8..15,24..31}
    AFrag a;
    a.q[0] = *(const uint4*)(arow + k0 + half * 8);
    a.q[1] = *(const uint4*)(arow + k0 + 16 + half * 8);
    // B 32x16 bf16 frags: col=l15; lanes<16 K={0..7,16..23}; lanes>=16 K={8..15,24..31}
    AFrag b[TN];
    #pragma unroll
    for (int t = 0; t < TN; t++) {
      b[t].q[0] = *(const uint4*)(brow[t] + k0 + half * 8);
      b[t].q[1] = *(const uint4*)(brow[t] + k0 + 16 + half * 8);
    }
    #pragma unroll
    for (int t = 0; t < TN; t++)
      c[t].v = __builtin_amdgcn_wmma_f32_16x16x32_bf16(
          false, a.v, false, b[t].v, (short)0, c[t].v, false, false);
  }
  // D layout: VGPR v -> row tm*16 + v + half*8
  int rowbase = tm * 16 + half * 8;
  #pragma unroll
  for (int t = 0; t < TN; t++) {
    int col = (tn0 + t) * 16 + l15;
    #pragma unroll
    for (int v = 0; v < 8; v++)
      C[(size_t)(rowbase + v) * N + col] = c[t].f[v];
  }
}

__global__ __launch_bounds__(256) void fill_u32_k(unsigned* p, unsigned v, size_t n) {
  size_t i = (size_t)blockIdx.x * blockDim.x + threadIdx.x;
  size_t stride = (size_t)gridDim.x * blockDim.x;
  for (; i < n; i += stride) p[i] = v;
}

// s[row] = dot(h[row,:], a[:]) — one wave per row, float4 lanes, shuffle reduce
__global__ __launch_bounds__(256) void rowdot_k(
    const float* __restrict__ h, const float* __restrict__ a,
    float* __restrict__ s, int M, int Nf) {
  int wid  = (int)((blockIdx.x * blockDim.x + threadIdx.x) >> 5);
  int lane = threadIdx.x & 31;
  if (wid >= M) return;
  const float4* r4 = (const float4*)(h + (size_t)wid * Nf);
  const float4* a4 = (const float4*)a;
  int nq = Nf >> 2;
  float acc = 0.0f;
  for (int k = lane; k < nq; k += 32) {
    float4 hv = r4[k], av = a4[k];
    acc += hv.x * av.x + hv.y * av.y + hv.z * av.z + hv.w * av.w;
  }
  #pragma unroll
  for (int o = 16; o > 0; o >>= 1) acc += __shfl_xor(acc, o, 32);
  if (lane == 0) s[wid] = acc;
}

__global__ __launch_bounds__(256) void edge_logit_k(
    const int* __restrict__ src, const int* __restrict__ dst,
    const float* __restrict__ ss, const float* __restrict__ sd,
    float* __restrict__ logit, unsigned* __restrict__ segmax, int ne) {
  int e = blockIdx.x * blockDim.x + threadIdx.x;
  if (e >= ne) return;
  float l = ss[src[e]] + sd[dst[e]];
  l = (l > 0.0f) ? l : 0.2f * l;             // leaky_relu
  logit[e] = l;
  atomicMax(&segmax[dst[e]], f2ord(l));
}

__global__ __launch_bounds__(256) void edge_exp_k(
    const int* __restrict__ dst, float* __restrict__ logit,
    const unsigned* __restrict__ segmax, float* __restrict__ den, int ne) {
  int e = blockIdx.x * blockDim.x + threadIdx.x;
  if (e >= ne) return;
  float m = ord2f(segmax[dst[e]]);
  if (!isfinite(m)) m = 0.0f;                 // empty segments
  float ex = __expf(logit[e] - m);
  logit[e] = ex;
  atomicAdd(&den[dst[e]], ex);
}

__global__ __launch_bounds__(256) void edge_alpha_k(
    const int* __restrict__ dst, float* __restrict__ logit,
    const float* __restrict__ den, int ne) {
  int e = blockIdx.x * blockDim.x + threadIdx.x;
  if (e >= ne) return;
  logit[e] = logit[e] / (den[dst[e]] + 1e-16f);
}

// out[dst[e], k..k+3] += hs[src[e], k..k+3] * alpha[e]
// One thread per (edge, 4 cols): one b128 gather + 4 f32 atomics.
__global__ __launch_bounds__(256) void edge_aggr_k(
    const int* __restrict__ src, const int* __restrict__ dst,
    const float* __restrict__ alpha, const float* __restrict__ hs,
    float* __restrict__ out, int ne, int Nf) {
  int nq = Nf >> 2;
  long long idx = (long long)blockIdx.x * blockDim.x + threadIdx.x;
  if (idx >= (long long)ne * nq) return;
  int e  = (int)(idx / nq);
  int k4 = (int)(idx - (long long)e * nq) << 2;
  float a = alpha[e];
  float4 h4 = *(const float4*)(hs + (size_t)src[e] * Nf + k4);
  float* o = out + (size_t)dst[e] * Nf + k4;
  atomicAdd(o + 0, h4.x * a);
  atomicAdd(o + 1, h4.y * a);
  atomicAdd(o + 2, h4.z * a);
  atomicAdd(o + 3, h4.w * a);
}

__global__ __launch_bounds__(256) void bias_act_k(
    const float* __restrict__ in, const float* __restrict__ b,
    float* __restrict__ out, int M, int Nf, int relu) {
  int nq = Nf >> 2;
  long long idx = (long long)blockIdx.x * blockDim.x + threadIdx.x;
  if (idx >= (long long)M * nq) return;
  int n4 = (int)(idx % nq) << 2;
  float4 v = ((const float4*)in)[idx];
  float4 bb = *(const float4*)(b + n4);
  v.x += bb.x; v.y += bb.y; v.z += bb.z; v.w += bb.w;
  if (relu) {
    v.x = fmaxf(v.x, 0.0f); v.y = fmaxf(v.y, 0.0f);
    v.z = fmaxf(v.z, 0.0f); v.w = fmaxf(v.w, 0.0f);
  }
  ((float4*)out)[idx] = v;
}

// s16[k] += xb[dst_idx[i], k]
__global__ __launch_bounds__(256) void sum_sel_k(
    const float* __restrict__ xb, const int* __restrict__ dsti,
    float* __restrict__ s16, int nl) {
  long long idx = (long long)blockIdx.x * blockDim.x + threadIdx.x;
  if (idx >= (long long)nl * O_) return;
  int i = (int)(idx >> 4), k = (int)(idx & 15);
  atomicAdd(&s16[k], xb[(size_t)dsti[i] * O_ + k]);
}

__global__ __launch_bounds__(256) void pred_k(
    const float* __restrict__ xa, const int* __restrict__ srci,
    const float* __restrict__ s16, float* __restrict__ pred, int nl) {
  int i = blockIdx.x * blockDim.x + threadIdx.x;
  if (i >= nl) return;
  const float* r = xa + (size_t)srci[i] * O_;
  float acc = 0.0f;
  #pragma unroll
  for (int k = 0; k < O_; k++) acc += r[k] * s16[k];
  pred[i] = acc;
}

static inline int cdiv_i(long long a, long long b) { return (int)((a + b - 1) / b); }

static void launch_gemm(const unsigned short* Abf, const unsigned short* Wt,
                        float* C, int M, int K, int N, hipStream_t stream) {
  if (N >= 64) {
    int waves = (M >> 4) * ((N >> 4) / 4);
    gat_wmma_gemm<4><<<cdiv_i((long long)waves * 32, 256), 256, 0, stream>>>(Abf, Wt, C, M, K, N);
  } else {
    int waves = (M >> 4) * (N >> 4);
    gat_wmma_gemm<1><<<cdiv_i((long long)waves * 32, 256), 256, 0, stream>>>(Abf, Wt, C, M, K, N);
  }
}

static void launch_conv(const unsigned short* xs_bf, const unsigned short* xd_bf, int Kin,
                        const int* src, const int* dst,
                        const float* W, const float* a_s, const float* a_d,
                        int Nf, unsigned short* wt_bf, float* h_src, float* h_dst,
                        float* logit, float* den, unsigned* segmax, float* ss, float* sd,
                        float* outbuf, hipStream_t stream) {
  transpose_bf16_k<<<cdiv_i((long long)Kin * Nf, 256), 256, 0, stream>>>(W, wt_bf, Kin, Nf);
  // hs = xs @ W ; hd = xd @ W (shared linear)
  launch_gemm(xs_bf, wt_bf, h_src, NA_, Kin, Nf, stream);
  launch_gemm(xd_bf, wt_bf, h_dst, NB_, Kin, Nf, stream);
  // attention coefficients per node
  rowdot_k<<<cdiv_i((long long)NA_ * 32, 256), 256, 0, stream>>>(h_src, a_s, ss, NA_, Nf);
  rowdot_k<<<cdiv_i((long long)NB_ * 32, 256), 256, 0, stream>>>(h_dst, a_d, sd, NB_, Nf);
  // segment softmax state
  fill_u32_k<<<1024, 256, 0, stream>>>(segmax, ORD_NEG_INF, (size_t)NB_);
  fill_u32_k<<<1024, 256, 0, stream>>>((unsigned*)den, 0u, (size_t)NB_);
  fill_u32_k<<<4096, 256, 0, stream>>>((unsigned*)outbuf, 0u, (size_t)NB_ * Nf);
  edge_logit_k<<<cdiv_i(E_, 256), 256, 0, stream>>>(src, dst, ss, sd, logit, segmax, E_);
  edge_exp_k<<<cdiv_i(E_, 256), 256, 0, stream>>>(dst, logit, segmax, den, E_);
  edge_alpha_k<<<cdiv_i(E_, 256), 256, 0, stream>>>(dst, logit, den, E_);
  edge_aggr_k<<<cdiv_i((long long)E_ * (Nf >> 2), 256), 256, 0, stream>>>(
      src, dst, logit, h_src, outbuf, E_, Nf);
}

extern "C" void kernel_launch(void* const* d_in, const int* in_sizes, int n_in,
                              void* d_out, int out_size, void* d_ws, size_t ws_size,
                              hipStream_t stream) {
  const float* xA = (const float*)d_in[0];
  const float* xB = (const float*)d_in[1];
  const int* ei_ab = (const int*)d_in[2];   // [2,E]: src row then dst row
  const int* ei_ba = (const int*)d_in[3];
  const int* src_idx = (const int*)d_in[4];
  const int* dst_idx = (const int*)d_in[5];
  float* out = (float*)d_out;               // pred[L] | xa[NA*16] | xb[NB*16]

  // workspace carve
  float* p = (float*)d_ws;
  float* xa_buf = p; p += (size_t)NA_ * H_;
  float* xb_buf = p; p += (size_t)NB_ * H_;
  float* h_src  = p; p += (size_t)NA_ * H_;
  float* h_dst  = p; p += (size_t)NB_ * H_;
  float* out_a  = p; p += (size_t)NA_ * H_;
  float* out_b  = p; p += (size_t)NB_ * H_;
  float* logit  = p; p += (size_t)E_;
  float* den    = p; p += (size_t)NA_;
  unsigned* segmax = (unsigned*)p; p += (size_t)NA_;
  float* ss     = p; p += (size_t)NA_;
  float* sd     = p; p += (size_t)NA_;
  float* s16    = p; p += 16;
  unsigned short* xa_bf = (unsigned short*)p; p += (size_t)NA_ * H_ / 2;
  unsigned short* xb_bf = (unsigned short*)p; p += (size_t)NB_ * H_ / 2;
  unsigned short* wt_bf = (unsigned short*)p; p += (size_t)H_ * H_ / 2;

  // d_out is poisoned; layer-3 aggregation atomically accumulates into it.
  fill_u32_k<<<2048, 256, 0, stream>>>((unsigned*)out, 0u, (size_t)out_size);

  const float* cur_a = xA;
  const float* cur_b = xB;
  int Kin = F0_;
  float* xa_final = out + L_;
  float* xb_final = out + L_ + (size_t)NA_ * O_;

  for (int l = 0; l < 3; l++) {
    int Nf = (l == 2) ? O_ : H_;
    const float* Wab  = (const float*)d_in[6 + l * 8 + 0];
    const float* asab = (const float*)d_in[6 + l * 8 + 1];
    const float* adab = (const float*)d_in[6 + l * 8 + 2];
    const float* bab  = (const float*)d_in[6 + l * 8 + 3];
    const float* Wba  = (const float*)d_in[6 + l * 8 + 4];
    const float* asba = (const float*)d_in[6 + l * 8 + 5];
    const float* adba = (const float*)d_in[6 + l * 8 + 6];
    const float* bba  = (const float*)d_in[6 + l * 8 + 7];
    float* oB = (l == 2) ? xb_final : out_b;
    float* oA = (l == 2) ? xa_final : out_a;

    // bf16 copies of current features (read many times by WMMA GEMMs)
    cast_bf16_k<<<4096, 256, 0, stream>>>(cur_a, (unsigned*)xa_bf, (size_t)NA_ * Kin / 2);
    cast_bf16_k<<<4096, 256, 0, stream>>>(cur_b, (unsigned*)xb_bf, (size_t)NB_ * Kin / 2);

    // A -> B conv (produces new xb side)
    launch_conv(xa_bf, xb_bf, Kin, ei_ab, ei_ab + E_, Wab, asab, adab, Nf,
                wt_bf, h_src, h_dst, logit, den, segmax, ss, sd, oB, stream);
    // B -> A conv (produces new xa side)
    launch_conv(xb_bf, xa_bf, Kin, ei_ba, ei_ba + E_, Wba, asba, adba, Nf,
                wt_bf, h_src, h_dst, logit, den, segmax, ss, sd, oA, stream);

    int relu = (l < 2) ? 1 : 0;
    float* nxt_a = (l == 2) ? xa_final : xa_buf;
    float* nxt_b = (l == 2) ? xb_final : xb_buf;
    bias_act_k<<<cdiv_i((long long)NA_ * (Nf >> 2), 256), 256, 0, stream>>>(oA, bba, nxt_a, NA_, Nf, relu);
    bias_act_k<<<cdiv_i((long long)NB_ * (Nf >> 2), 256), 256, 0, stream>>>(oB, bab, nxt_b, NB_, Nf, relu);
    cur_a = nxt_a; cur_b = nxt_b; Kin = Nf;
  }

  // pred = xa[src_idx] @ xb[dst_idx].sum(0)
  fill_u32_k<<<1, 256, 0, stream>>>((unsigned*)s16, 0u, (size_t)16);
  sum_sel_k<<<cdiv_i((long long)L_ * O_, 256), 256, 0, stream>>>(xb_final, dst_idx, s16, L_);
  pred_k<<<cdiv_i(L_, 256), 256, 0, stream>>>(xa_final, src_idx, s16, out, L_);
}
